// AdditiveAttention_2061584302494
// MI455X (gfx1250) — matrix operand order repaired and checked by reading, not compiled, not run on hardware
//
#include <hip/hip_runtime.h>

// ---- Problem constants (AdditiveAttention) ----
#define B_DIM   64
#define S_LEN   2048
#define E_DIM   512      // ENC_H == DEC_H
#define A_DIM   512      // ATTN

// ---- WMMA vector types (gfx1250, wave32) ----
typedef __attribute__((ext_vector_type(16))) __bf16 v16bf;
typedef __attribute__((ext_vector_type(8)))  __bf16 v8bf;
typedef __attribute__((ext_vector_type(4)))  __bf16 v4bf;
typedef __attribute__((ext_vector_type(8)))  float  v8f;

#define LOG2E 1.44269504088896340736f

__device__ __forceinline__ float fast_tanh(float x) {
    // tanh(x) = 1 - 2/(1+exp(2x)); exp(2x) = exp2(2*log2e*x)
    float e = __builtin_amdgcn_exp2f(x * (2.0f * LOG2E));
    return 1.0f - 2.0f * __builtin_amdgcn_rcpf(1.0f + e);
}

// ============================================================
// Fragment block format (shared by A and B operands), per (tile,k):
//   1 KB = 512 bf16:
//     bytes [lane*16 , +16)        = that lane's first 8-element K chunk
//     bytes [512 + lane*16 , +16)  = that lane's second 8-element K chunk
// ds_load_b128 across 32 lanes -> every bank exactly twice (optimal).
// ============================================================

// ============================================================
// Kernel 0: pack W_enc (fp32 [A,E] row-major) into bf16 B-fragment blocks.
// Block (n,k): col a = n*16+l, K = e = k*32 + 16*hi + (0..15).
// ============================================================
__global__ __launch_bounds__(256)
void pack_wenc_kernel(const float* __restrict__ Wenc, unsigned short* __restrict__ Wpack_) {
    __bf16* Wpack = reinterpret_cast<__bf16*>(Wpack_);
    int t    = blockIdx.x * 256 + threadIdx.x;   // 16384 = 32n * 16k * 32lane
    int lane = t & 31;
    int k    = (t >> 5) & 15;
    int n    = t >> 9;
    int l    = lane & 15, hi = lane >> 4;
    int a    = n * 16 + l;
    int e    = k * 32 + 16 * hi;
    const float4* src = (const float4*)(Wenc + (size_t)a * E_DIM + e);
    float4 f0 = src[0], f1 = src[1], f2 = src[2], f3 = src[3];
    v8bf c0, c1;
    c0[0] = (__bf16)f0.x; c0[1] = (__bf16)f0.y; c0[2] = (__bf16)f0.z; c0[3] = (__bf16)f0.w;
    c0[4] = (__bf16)f1.x; c0[5] = (__bf16)f1.y; c0[6] = (__bf16)f1.z; c0[7] = (__bf16)f1.w;
    c1[0] = (__bf16)f2.x; c1[1] = (__bf16)f2.y; c1[2] = (__bf16)f2.z; c1[3] = (__bf16)f2.w;
    c1[4] = (__bf16)f3.x; c1[5] = (__bf16)f3.y; c1[6] = (__bf16)f3.z; c1[7] = (__bf16)f3.w;
    size_t blk = (size_t)(n * 16 + k) * 512;       // 512 bf16 = 1 KB per fragment block
    *(v8bf*)&Wpack[blk + lane * 8]       = c0;
    *(v8bf*)&Wpack[blk + 256 + lane * 8] = c1;
}

// ============================================================
// Kernel 1: dec_proj[b,a] = dot(dec_hidden[b,:], W_dec[a,:])   (tiny)
// ============================================================
__global__ __launch_bounds__(256)
void dproj_kernel(const float* __restrict__ dec, const float* __restrict__ Wdec,
                  float* __restrict__ dproj) {
    int idx = blockIdx.x * 256 + threadIdx.x;   // 64*512
    int b = idx >> 9, a = idx & 511;
    const float4* dp = (const float4*)(dec + (size_t)b * E_DIM);
    const float4* wp = (const float4*)(Wdec + (size_t)a * E_DIM);
    float s = 0.f;
    for (int i = 0; i < E_DIM / 4; ++i) {
        float4 x = dp[i], y = wp[i];
        s += x.x * y.x + x.y * y.y + x.z * y.z + x.w * y.w;
    }
    dproj[idx] = s;
}

// ============================================================
// Kernel 2 (main): fused  score[b,s] = sum_a tanh(enc@W_enc^T + dproj) * v
// Block = 256 threads (8 waves, 2/SIMD), block tile M = 256 S-rows.
// Each wave owns M=32 (two 16-row A tiles sharing every B fragment).
// A staged fragment-packed in LDS (256 KB); B chunk (4 col-tiles x full K,
// 64 KB) staged per nb step -> LDS total = 320 KB (full CDNA5 WG budget).
// Inner loop: per k-step 4 ds_b128 (A) + 8 ds_b128 (B) feed 8 WMMAs;
// zero global traffic.
// ============================================================
#define MBLK     256    // S-rows per block (8 waves x 32)
#define NB_CHUNK 4      // column tiles staged per step (64 cols)

__global__ __launch_bounds__(256)
void score_kernel(const float* __restrict__ enc, const unsigned short* __restrict__ Wpack_,
                  const float* __restrict__ dproj, const float* __restrict__ vvec,
                  float* __restrict__ scores) {
    const __bf16* Wpack = reinterpret_cast<const __bf16*>(Wpack_);
    __shared__ __bf16 ablk[(MBLK / 16) * 16 * 512];       // 262,144 B (frag-packed A)
    __shared__ __bf16 bbuf[NB_CHUNK * 16 * 512];          //  65,536 B

    const int b   = blockIdx.x >> 3;
    const int s0  = (blockIdx.x & 7) * MBLK;
    const int tid = threadIdx.x;

    // ---- Stage enc[b, s0+tid, :] into A fragment blocks (row per thread) ----
    // chunk c (8 e-values): k = c>>2, m = c&3 -> hi = m>>1, half q = m&1
    {
        const int tile = tid >> 4, l = tid & 15;
        const float4* rsrc = (const float4*)(enc + ((size_t)b * S_LEN + s0 + tid) * E_DIM);
        for (int c = 0; c < 64; ++c) {
            float4 f0 = rsrc[c * 2], f1 = rsrc[c * 2 + 1];
            v8bf h;
            h[0] = (__bf16)f0.x; h[1] = (__bf16)f0.y; h[2] = (__bf16)f0.z; h[3] = (__bf16)f0.w;
            h[4] = (__bf16)f1.x; h[5] = (__bf16)f1.y; h[6] = (__bf16)f1.z; h[7] = (__bf16)f1.w;
            int k = c >> 2, hi = (c >> 1) & 1, q = c & 1;
            *(v8bf*)&ablk[(size_t)(tile * 16 + k) * 512 + q * 256 + (hi * 16 + l) * 8] = h;
        }
    }

    const int wave = tid >> 5;
    const int lane = tid & 31;
    const int l    = lane & 15;
    const int hi   = lane >> 4;
    const int t0   = wave * 2;          // this wave's two 16-row A tiles

    float srow[2][8];
#pragma unroll
    for (int t = 0; t < 2; ++t)
#pragma unroll
        for (int r = 0; r < 8; ++r) srow[t][r] = 0.f;

    const v8f vzero = {0.f, 0.f, 0.f, 0.f, 0.f, 0.f, 0.f, 0.f};

    for (int nb = 0; nb < A_DIM / (NB_CHUNK * 16); ++nb) {   // 8 steps of 64 columns
        // ---- Stage 64 KB B chunk (contiguous in Wpack) into LDS ----
        __syncthreads();   // previous chunk consumed / A staging done
        {
            const uint4* src = (const uint4*)(Wpack + (size_t)nb * (NB_CHUNK * 16 * 512));
            uint4* dst = (uint4*)bbuf;
#pragma unroll
            for (int j = 0; j < 16; ++j)                     // 4096 uint4 / 256 threads
                dst[tid + j * 256] = src[tid + j * 256];
            if (nb + 1 < A_DIM / (NB_CHUNK * 16)) {          // prefetch next chunk -> L2
                const char* pf = (const char*)(Wpack + (size_t)(nb + 1) * (NB_CHUNK * 16 * 512));
                __builtin_prefetch(pf + tid * 256, 0, 1);
            }
        }
        __syncthreads();

        v8f acc[2][NB_CHUNK];
#pragma unroll
        for (int t = 0; t < 2; ++t)
#pragma unroll
            for (int c = 0; c < NB_CHUNK; ++c) acc[t][c] = vzero;

        for (int k = 0; k < E_DIM / 32; ++k) {               // K = 512 in steps of 32
            union { v16bf v; v8bf h[2]; } af0, af1;
            {
                size_t u0 = (size_t)((t0 + 0) * 16 + k) * 512;
                size_t u1 = (size_t)((t0 + 1) * 16 + k) * 512;
                af0.h[0] = *(const v8bf*)&ablk[u0 + lane * 8];
                af0.h[1] = *(const v8bf*)&ablk[u0 + 256 + lane * 8];
                af1.h[0] = *(const v8bf*)&ablk[u1 + lane * 8];
                af1.h[1] = *(const v8bf*)&ablk[u1 + 256 + lane * 8];
            }
#pragma unroll
            for (int c = 0; c < NB_CHUNK; ++c) {
                size_t u = (size_t)(c * 16 + k) * 512;       // split-half fragment block
                union { v16bf v; v8bf h[2]; } bf_;
                bf_.h[0] = *(const v8bf*)&bbuf[u + lane * 8];
                bf_.h[1] = *(const v8bf*)&bbuf[u + 256 + lane * 8];
                acc[0][c] = __builtin_amdgcn_wmma_f32_16x16x32_bf16(
                    false, af0.v, false, bf_.v, (short)0, acc[0][c], false, false);
                acc[1][c] = __builtin_amdgcn_wmma_f32_16x16x32_bf16(
                    false, af1.v, false, bf_.v, (short)0, acc[1][c], false, false);
            }
        }
        // ---- fused epilogue: tanh(P + dproj) * v, accumulate per-row ----
#pragma unroll
        for (int c = 0; c < NB_CHUNK; ++c) {
            int a = (nb * NB_CHUNK + c) * 16 + l;            // column this lane owns
            float dp = dproj[(size_t)b * A_DIM + a];
            float vv = vvec[a];
#pragma unroll
            for (int t = 0; t < 2; ++t)
#pragma unroll
                for (int r = 0; r < 8; ++r) {                // C/D: VGPR r <-> row r+8*hi
                    float x = fast_tanh(acc[t][c][r] + dp);
                    srow[t][r] += x * vv;
                }
        }
    }

    // ---- reduce over the 16 lanes (columns) of each half-wave ----
#pragma unroll
    for (int t = 0; t < 2; ++t)
#pragma unroll
        for (int r = 0; r < 8; ++r)
            for (int off = 1; off < 16; off <<= 1)
                srow[t][r] += __shfl_xor(srow[t][r], off, 32);

    if (l == 0) {                                   // lanes 0 & 16 write rows 0-7 / 8-15
#pragma unroll
        for (int t = 0; t < 2; ++t) {
            float* sp = scores + (size_t)b * S_LEN + s0 + wave * 32 + t * 16 + hi * 8;
#pragma unroll
            for (int r = 0; r < 8; ++r) sp[r] = srow[t][r];
        }
    }
}

// ============================================================
// Kernel 3: softmax over S=2048 per batch row.
// ============================================================
__global__ __launch_bounds__(256)
void softmax_kernel(const float* __restrict__ scores, float* __restrict__ attn) {
    __shared__ float red[256];
    int b = blockIdx.x, tid = threadIdx.x;
    const float* sp = scores + (size_t)b * S_LEN;

    float sv[8], mx = -3.4e38f;
#pragma unroll
    for (int i = 0; i < 8; ++i) { sv[i] = sp[i * 256 + tid]; mx = fmaxf(mx, sv[i]); }
    red[tid] = mx; __syncthreads();
    for (int o = 128; o > 0; o >>= 1) {
        if (tid < o) red[tid] = fmaxf(red[tid], red[tid + o]);
        __syncthreads();
    }
    float bm = red[0]; __syncthreads();

    float ex[8], ls = 0.f;
#pragma unroll
    for (int i = 0; i < 8; ++i) { ex[i] = __builtin_amdgcn_exp2f((sv[i] - bm) * LOG2E); ls += ex[i]; }
    red[tid] = ls; __syncthreads();
    for (int o = 128; o > 0; o >>= 1) {
        if (tid < o) red[tid] += red[tid + o];
        __syncthreads();
    }
    float inv = 1.0f / red[0];
#pragma unroll
    for (int i = 0; i < 8; ++i) attn[(size_t)b * S_LEN + i * 256 + tid] = ex[i] * inv;
}

// ============================================================
// Kernel 4: context partials over 16 S-chunks (deterministic; no fp atomics).
// grid = (16 chunks, 64 b), block = 256; thread owns 2 enc columns.
// ============================================================
__global__ __launch_bounds__(256)
void context_partial_kernel(const float* __restrict__ enc, const float* __restrict__ attn,
                            float* __restrict__ partial) {
    int chunk = blockIdx.x, b = blockIdx.y;
    int tid = threadIdx.x;
    const float2* ep = (const float2*)(enc + ((size_t)b * S_LEN + chunk * 128) * E_DIM);
    const float*  wp = attn + (size_t)b * S_LEN + chunk * 128;
    float2 acc = {0.f, 0.f};
    for (int s = 0; s < 128; ++s) {
        float w = wp[s];
        float2 x = ep[(size_t)s * (E_DIM / 2) + tid];
        acc.x += w * x.x;
        acc.y += w * x.y;
    }
    ((float2*)(partial + ((size_t)(b * 16 + chunk)) * E_DIM))[tid] = acc;
}

// ============================================================
// Kernel 5: reduce 16 partials -> context[b,e].
// ============================================================
__global__ __launch_bounds__(512)
void context_reduce_kernel(const float* __restrict__ partial, float* __restrict__ ctx) {
    int b = blockIdx.x, e = threadIdx.x;
    float s = 0.f;
#pragma unroll
    for (int c = 0; c < 16; ++c) s += partial[((size_t)(b * 16 + c)) * E_DIM + e];
    ctx[(size_t)b * E_DIM + e] = s;
}

// ============================================================
// Host launcher
// ============================================================
extern "C" void kernel_launch(void* const* d_in, const int* in_sizes, int n_in,
                              void* d_out, int out_size, void* d_ws, size_t ws_size,
                              hipStream_t stream) {
    const float* enc   = (const float*)d_in[0];   // [64,2048,512]
    const float* dec   = (const float*)d_in[1];   // [64,512]
    const float* Wenc  = (const float*)d_in[2];   // [512,512]
    const float* Wdec  = (const float*)d_in[3];   // [512,512]
    const float* vvec  = (const float*)d_in[4];   // [1,512]

    float* ctx  = (float*)d_out;                  // [64,512]
    float* attn = ctx + (size_t)B_DIM * E_DIM;    // [64,2048]

    char* base = (char*)d_ws;
    unsigned short* Wpack = (unsigned short*)(base);                       // 512 KB (bf16)
    float* dproj   = (float*)(base + 524288);                              // 128 KB
    float* scores  = (float*)(base + 524288 + 131072);                     // 512 KB
    float* partial = (float*)(base + 524288 + 131072 + 524288);            // 2 MB

    pack_wenc_kernel<<<64, 256, 0, stream>>>(Wenc, Wpack);
    dproj_kernel<<<128, 256, 0, stream>>>(dec, Wdec, dproj);
    score_kernel<<<B_DIM * (S_LEN / MBLK), 256, 0, stream>>>(enc, Wpack, dproj, vvec, scores);
    softmax_kernel<<<B_DIM, 256, 0, stream>>>(scores, attn);
    context_partial_kernel<<<dim3(16, B_DIM), 256, 0, stream>>>(enc, attn, partial);
    context_reduce_kernel<<<B_DIM, 512, 0, stream>>>(partial, ctx);
}
